// GIN_80728205295658
// MI455X (gfx1250) — compile-verified, matrix operand order checked by reading, not run on hardware
//
#include <hip/hip_runtime.h>
#include <stdint.h>

// ---------------- Types for WMMA ----------------
typedef __bf16 bf16_t;
typedef bf16_t v16bf __attribute__((ext_vector_type(16)));
typedef float  v8f   __attribute__((ext_vector_type(8)));

__device__ __forceinline__ unsigned short f2bf(float f) {
    union { float f; unsigned u; } x; x.f = f;
    unsigned r = x.u + 0x7FFFu + ((x.u >> 16) & 1u);   // round-to-nearest-even
    return (unsigned short)(r >> 16);
}

// ---------------- Elementwise helpers ----------------
__global__ void k_zero(float* __restrict__ p, long long n) {
    long long i = (long long)blockIdx.x * blockDim.x + threadIdx.x;
    if (i < n) p[i] = 0.0f;
}

__global__ void k_copy4(float4* __restrict__ dst, const float4* __restrict__ src, long long n4) {
    long long i = (long long)blockIdx.x * blockDim.x + threadIdx.x;
    if (i < n4) dst[i] = src[i];
}

// Wt[n][k] = bf16(W[k][n]), zero-padded rows for n in [N, NPAD)
__global__ void k_wt(const float* __restrict__ W, unsigned short* __restrict__ Wt,
                     int N, int K, int logK, long long total) {
    long long i = (long long)blockIdx.x * blockDim.x + threadIdx.x;
    if (i >= total) return;
    int k = (int)(i & (long long)(K - 1));
    int n = (int)(i >> logK);
    Wt[i] = (n < N) ? f2bf(W[(size_t)k * N + n]) : (unsigned short)0;
}

// out[dst[e]*F + f] += x[src[e]*F + f]   (F4 = F/4, power of two)
__global__ void k_edge_agg4(const long long* __restrict__ src, const long long* __restrict__ dst,
                            const float4* __restrict__ x, float* __restrict__ out,
                            long long E, int F4, int logF4) {
    long long tid = (long long)blockIdx.x * blockDim.x + threadIdx.x;
    if (tid >= E * (long long)F4) return;
    long long e = tid >> logF4;
    int f = (int)(tid & (long long)(F4 - 1));
    long long s = src[e];
    long long d = dst[e];
    float4 v = x[s * (long long)F4 + f];
    float* o = out + ((d * (long long)F4 + f) << 2);
    atomicAdd(o + 0, v.x);
    atomicAdd(o + 1, v.y);
    atomicAdd(o + 2, v.z);
    atomicAdd(o + 3, v.w);
}

// Column sums / sums-of-squares for BN over M rows x 512 cols.
// stats[0..511] = sum, stats[512..1023] = sumsq.
__global__ __launch_bounds__(256) void k_colstats(const float* __restrict__ z,
                                                  float* __restrict__ stats, int M) {
    int c0 = threadIdx.x;            // 0..255
    int r0 = blockIdx.x * 128;
    int rend = r0 + 128; if (rend > M) rend = M;
    float s0 = 0.f, s1 = 0.f, q0 = 0.f, q1 = 0.f;
    for (int r = r0; r < rend; ++r) {
        float a = z[(size_t)r * 512 + c0];
        float b = z[(size_t)r * 512 + c0 + 256];
        s0 += a; q0 += a * a;
        s1 += b; q1 += b * b;
    }
    atomicAdd(&stats[c0], s0);
    atomicAdd(&stats[c0 + 256], s1);
    atomicAdd(&stats[512 + c0], q0);
    atomicAdd(&stats[512 + c0 + 256], q1);
}

// In-place BN (training stats, biased var) + ReLU, 512 channels.
__global__ void k_bn_relu(float* __restrict__ z, const float* __restrict__ stats,
                          const float* __restrict__ gamma, const float* __restrict__ beta, int M) {
    long long i = (long long)blockIdx.x * blockDim.x + threadIdx.x;
    if (i >= (long long)M * 512) return;
    int c = (int)(i & 511);
    float invM = 1.0f / (float)M;
    float mu  = stats[c] * invM;
    float var = stats[512 + c] * invM - mu * mu;
    float y = gamma[c] * (z[i] - mu) * rsqrtf(var + 1e-5f) + beta[c];
    z[i] = fmaxf(y, 0.0f);
}

// pooled[batch[row]][c] += h[row][c], 512 channels
__global__ void k_pool(const float* __restrict__ h, const long long* __restrict__ batch,
                       float* __restrict__ pooled, int M) {
    long long i = (long long)blockIdx.x * blockDim.x + threadIdx.x;
    if (i >= (long long)M * 512) return;
    int c = (int)(i & 511);
    long long row = i >> 9;
    long long g = batch[row];
    atomicAdd(&pooled[g * 512 + c], h[i]);
}

// ---------------- WMMA GEMM: C = act(A[MxK] * Wt^T + bias) ----------------
// Wt is bf16, N-major [NPAD][K] (zero-padded rows), L2-resident.
// Block tile 128(M) x 128(N): 8 waves as 4(M) x 2(N); each wave computes 32x64
// = 2 A-fragments x 4 B-fragments -> 8 WMMAs per 32-wide k-step.
// A tile goes through double-buffered LDS with f32->bf16 conversion on store.
// B fragments load straight from global Wt (two b128 per lane per fragment).
// Requires K % 32 == 0, K power-of-two-friendly sizes (128/512 here).
#define BM 128
#define BNB 128
#define BK 32
#define LDA 34

__global__ __launch_bounds__(256)
void k_gemm_bias_act(const float* __restrict__ A, const unsigned short* __restrict__ Wt,
                     const float* __restrict__ bias, float* __restrict__ C,
                     int M, int N, int K, int relu) {
    __shared__ __align__(16) unsigned short As[2][BM * LDA];  // bf16 bits

    const int tid  = threadIdx.x;
    const int lane = tid & 31;
    const int wave  = tid >> 5;        // 0..7
    const int waveM = wave & 3;        // 0..3  (32 rows each)
    const int waveN = wave >> 2;       // 0..1  (64 cols each)
    const int m0 = blockIdx.y * BM;
    const int n0 = blockIdx.x * BNB;

    v8f acc[2][4];
    #pragma unroll
    for (int s = 0; s < 2; ++s)
        #pragma unroll
        for (int t = 0; t < 4; ++t)
            #pragma unroll
            for (int j = 0; j < 8; ++j) acc[s][t][j] = 0.0f;

    const int rA = tid >> 1;           // 0..127 (A tile row)
    const int cA = (tid & 1) * 16;     // 0 or 16
    const int halfsel = lane >> 4;     // 0: lanes 0-15, 1: lanes 16-31
    const int base = halfsel << 3;     // k sub-offset 0 or 8
    const int nfrag = (lane & 15);     // fragment column / row index

    union AFrag { unsigned u[8]; v16bf v; };
    union BFrag { uint4 q[2]; v16bf v; };

    unsigned packA[8];
    const int gm = m0 + rA;

    // ---- staged A tile load: global f32 -> packed bf16 pairs in regs ----
    auto loadA = [&](int k0) {
        if (gm < M) {
            const float4* p = (const float4*)(A + (size_t)gm * K + k0 + cA);
            #pragma unroll
            for (int j4 = 0; j4 < 4; ++j4) {
                float4 vv = p[j4];
                packA[j4 * 2 + 0] = (unsigned)f2bf(vv.x) | ((unsigned)f2bf(vv.y) << 16);
                packA[j4 * 2 + 1] = (unsigned)f2bf(vv.z) | ((unsigned)f2bf(vv.w) << 16);
            }
        } else {
            #pragma unroll
            for (int j = 0; j < 8; ++j) packA[j] = 0u;
        }
    };
    auto storeA = [&](int buf) {
        unsigned* dp = (unsigned*)&As[buf][rA * LDA + cA];   // 4B-aligned (rA*34+cA even)
        #pragma unroll
        for (int j = 0; j < 8; ++j) dp[j] = packA[j];
    };

    const int nk = K / BK;
    loadA(0);
    storeA(0);
    __syncthreads();

    for (int ks = 0; ks < nk; ++ks) {
        const int cur = ks & 1;
        if (ks + 1 < nk) loadA((ks + 1) * BK);

        // A fragments from LDS (16-bit A 16x32 layout, CDNA5 ISA 7.12.2)
        AFrag aF[2];
        #pragma unroll
        for (int sub = 0; sub < 2; ++sub) {
            const int ml = waveM * 32 + sub * 16 + nfrag;
            #pragma unroll
            for (int q = 0; q < 8; ++q) {
                int g = q >> 2, w = q & 3;
                int kk = g * 16 + base + 2 * w;
                aF[sub].u[q] = *(const unsigned*)&As[cur][ml * LDA + kk];
            }
        }
        // B fragments straight from global Wt (n-major, K-pairs contiguous)
        #pragma unroll
        for (int t = 0; t < 4; ++t) {
            BFrag bF;
            const int n = n0 + waveN * 64 + t * 16 + nfrag;
            const unsigned short* wp = Wt + (size_t)n * K + ks * BK + base;
            bF.q[0] = *(const uint4*)(wp);
            bF.q[1] = *(const uint4*)(wp + 16);
            acc[0][t] = __builtin_amdgcn_wmma_f32_16x16x32_bf16(
                false, aF[0].v, false, bF.v, (short)0, acc[0][t], false, false);
            acc[1][t] = __builtin_amdgcn_wmma_f32_16x16x32_bf16(
                false, aF[1].v, false, bF.v, (short)0, acc[1][t], false, false);
        }

        if (ks + 1 < nk) storeA(cur ^ 1);
        __syncthreads();
    }

    // ---- epilogue: VGPR r -> row (lanes0-15: r, lanes16-31: r+8), col = lane&15 ----
    #pragma unroll
    for (int sub = 0; sub < 2; ++sub) {
        const int mbase = m0 + waveM * 32 + sub * 16 + (halfsel << 3);
        #pragma unroll
        for (int t = 0; t < 4; ++t) {
            int n = n0 + waveN * 64 + t * 16 + nfrag;
            if (n >= N) continue;
            float bv = bias[n];
            #pragma unroll
            for (int r = 0; r < 8; ++r) {
                int m = mbase + r;
                if (m < M) {
                    float val = acc[sub][t][r] + bv;
                    if (relu) val = fmaxf(val, 0.0f);
                    C[(size_t)m * N + n] = val;
                }
            }
        }
    }
}

// ---------------- Host orchestration ----------------
static inline int nblk(long long n, int b) { return (int)((n + b - 1) / b); }

extern "C" void kernel_launch(void* const* d_in, const int* in_sizes, int n_in,
                              void* d_out, int out_size, void* d_ws, size_t ws_size,
                              hipStream_t stream) {
    const float*     x    = (const float*)d_in[0];
    const long long* ei   = (const long long*)d_in[1];
    const long long* bat  = (const long long*)d_in[2];
    const float *W1a = (const float*)d_in[3],  *b1a = (const float*)d_in[4];
    const float *g1  = (const float*)d_in[5],  *be1 = (const float*)d_in[6];
    const float *W1b = (const float*)d_in[7],  *b1b = (const float*)d_in[8];
    const float *W2a = (const float*)d_in[9],  *b2a = (const float*)d_in[10];
    const float *g2  = (const float*)d_in[11], *be2 = (const float*)d_in[12];
    const float *W2b = (const float*)d_in[13], *b2b = (const float*)d_in[14];
    const float *Wl1 = (const float*)d_in[15], *bl1 = (const float*)d_in[16];
    const float *Wl2 = (const float*)d_in[17], *bl2 = (const float*)d_in[18];

    const long long Nn = in_sizes[0] / 128;    // 50000 nodes
    const long long E  = in_sizes[1] / 2;      // 800000 edges
    const long long* src = ei;
    const long long* dst = ei + E;
    const int D = 512, IN = 128, G = 64, OUT = 10;
    const int LOG_IN = 7, LOG_D = 9;

    float* bufA  = (float*)d_ws;                     // Nn*512 f32
    float* bufB  = bufA + (size_t)Nn * D;            // Nn*512 f32
    float* stats = bufB + (size_t)Nn * D;            // 1024 f32
    float* P     = stats + 1024;                     // 64*512 pooled
    float* Q     = P + (size_t)G * D;                // 64*512 hidden
    unsigned short* Wt = (unsigned short*)(Q + (size_t)G * D);  // up to 512*512 bf16
    (void)ws_size; (void)n_in; (void)out_size;

    const int TB = 256;
    dim3 blk(TB);

    // helper lambdas for launches
    auto gemm = [&](const float* Ap, const float* Wp, const float* bp, float* Cp,
                    int M, int N, int K, int logK, int relu) {
        int NPAD = ((N + BNB - 1) / BNB) * BNB;
        long long total = (long long)NPAD * K;
        k_wt<<<nblk(total, TB), blk, 0, stream>>>(Wp, Wt, N, K, logK, total);
        dim3 grid(NPAD / BNB, (M + BM - 1) / BM);
        k_gemm_bias_act<<<grid, blk, 0, stream>>>(Ap, Wt, bp, Cp, M, N, K, relu);
    };

    // ---------- Layer 1 ----------
    // t1 = x + scatter_sum(x[src] -> dst)            (Nn x 128, in bufA)
    k_copy4<<<nblk(Nn * IN / 4, TB), blk, 0, stream>>>((float4*)bufA, (const float4*)x, Nn * IN / 4);
    k_edge_agg4<<<nblk(E * (IN / 4), TB), blk, 0, stream>>>(src, dst, (const float4*)x, bufA, E, IN / 4, 5);
    // z1 = t1 @ W1a + b1a                            (Nn x 512, in bufB)
    gemm(bufA, W1a, b1a, bufB, (int)Nn, D, IN, LOG_IN, 0);
    // BN + ReLU (in-place bufB)
    k_zero<<<nblk(1024, TB), blk, 0, stream>>>(stats, 1024);
    k_colstats<<<nblk(Nn, 128), blk, 0, stream>>>(bufB, stats, (int)Nn);
    k_bn_relu<<<nblk(Nn * D, TB), blk, 0, stream>>>(bufB, stats, g1, be1, (int)Nn);
    // h1 = relu(bn @ W1b + b1b)                      (Nn x 512, in bufA)
    gemm(bufB, W1b, b1b, bufA, (int)Nn, D, D, LOG_D, 1);

    // ---------- Layer 2 ----------
    // t2 = h1 + scatter_sum(h1[src] -> dst)          (bufB)
    k_copy4<<<nblk(Nn * D / 4, TB), blk, 0, stream>>>((float4*)bufB, (const float4*)bufA, Nn * D / 4);
    k_edge_agg4<<<nblk(E * (D / 4), TB), blk, 0, stream>>>(src, dst, (const float4*)bufA, bufB, E, D / 4, 7);
    // z2 = t2 @ W2a + b2a                            (bufA)
    gemm(bufB, W2a, b2a, bufA, (int)Nn, D, D, LOG_D, 0);
    k_zero<<<nblk(1024, TB), blk, 0, stream>>>(stats, 1024);
    k_colstats<<<nblk(Nn, 128), blk, 0, stream>>>(bufA, stats, (int)Nn);
    k_bn_relu<<<nblk(Nn * D, TB), blk, 0, stream>>>(bufA, stats, g2, be2, (int)Nn);
    // h2 = relu(bn @ W2b + b2b)                      (bufB)
    gemm(bufA, W2b, b2b, bufB, (int)Nn, D, D, LOG_D, 1);

    // ---------- Pool + MLP head ----------
    k_zero<<<nblk((long long)G * D, TB), blk, 0, stream>>>(P, (long long)G * D);
    k_pool<<<nblk(Nn * D, TB), blk, 0, stream>>>(bufB, bat, P, (int)Nn);
    gemm(P, Wl1, bl1, Q, G, D, D, LOG_D, 1);
    gemm(Q, Wl2, bl2, (float*)d_out, G, OUT, D, LOG_D, 0);
}
